// SGA_CostAggregation_58780922413131
// MI455X (gfx1250) — matrix operand order repaired and checked by reading, not compiled, not run on hardware
//
#include <hip/hip_runtime.h>
#include <hip/hip_bf16.h>

// ---------------------------------------------------------------------------
// Types for CDNA5 WMMA (wave32): A/B = 16 bf16 per lane (8 VGPRs),
// C/D = 8 f32 per lane (8 VGPRs).
// ---------------------------------------------------------------------------
typedef __bf16 v16bf __attribute__((ext_vector_type(16)));
typedef float  v8f   __attribute__((ext_vector_type(8)));

__device__ __forceinline__ __bf16 f2bf(float f) {
  unsigned u = __float_as_uint(f);
  unsigned r = (u + 0x7FFFu + ((u >> 16) & 1u)) >> 16;   // round-to-nearest-even
  unsigned short s = (unsigned short)r;
  return __builtin_bit_cast(__bf16, s);
}
__device__ __forceinline__ __bf16 bfz() {
  return __builtin_bit_cast(__bf16, (unsigned short)0);
}
__device__ __forceinline__ unsigned short bf_bits(__bf16 v) {
  return __builtin_bit_cast(unsigned short, v);
}

// ---------------------------------------------------------------------------
// fp32 -> bf16 streaming convert (activations / weights)
// ---------------------------------------------------------------------------
__global__ void k_tobf(const float* __restrict__ in, __bf16* __restrict__ out, long n) {
  long i = (long)blockIdx.x * blockDim.x + threadIdx.x;
  if (i < n) out[i] = f2bf(in[i]);
}

__global__ void k_relu(float* __restrict__ x, long n) {
  long i = (long)blockIdx.x * blockDim.x + threadIdx.x;
  if (i < n) x[i] = fmaxf(x[i], 0.f);
}

// ---------------------------------------------------------------------------
// Implicit-GEMM 3D conv / transposed-conv, bf16 in, fp32 out, via
// v_wmma_f32_16x16x32_bf16.
//   block tile : M=16 (Cout) x N=512 (output voxels), K = Cin*27 (padded to 32)
//   wave tile  : 16M x 64N  -> 4 accumulators, shared A fragment, 4 WMMAs/step
//   LDS        : double-buffered weight (16x32) + im2col (32x512) tiles,
//                one barrier per K-step (stage next while computing current)
//   normal :  id = od*stride + kd - 1            (pad 1)
//   deconv :  td = od + kd - 1; valid iff td even; id = td/2
//             (== lhs_dilation 2, pad (1,2), k=3 -> out = 2*in)
// ---------------------------------------------------------------------------
#define NB   512
#define LPITCH 520   // 520 bf16 = 1040 B = 65*16 B -> rows stay 16B-aligned

__global__ __launch_bounds__(256) void k_conv_igemm(
    const __bf16* __restrict__ act, const __bf16* __restrict__ wgt,
    float* __restrict__ out,
    int Cin, int Din, int Hin, int Win,
    int Dout, int Hout, int Wout, int stride, int deconv)
{
  const int K = Cin * 27;
  __shared__ __bf16 sI[2][32 * LPITCH];
  __shared__ __bf16 sW[2][16 * 32];

  const int t    = threadIdx.x;
  const int lane = t & 31;
  const int wv   = t >> 5;              // wave id 0..7
  const long NS  = (long)Dout * Hout * Wout;
  const long nblock = (long)blockIdx.x * NB;
  const int  mblock = blockIdx.y * 16;

  // this thread's two (adjacent) output columns
  int od[2], oh[2], ow[2];
#pragma unroll
  for (int c = 0; c < 2; ++c) {
    long n  = nblock + 2 * t + c;
    ow[c]   = (int)(n % Wout);
    long n2 = n / Wout;
    oh[c]   = (int)(n2 % Hout);
    od[c]   = (int)(n2 / Hout);
  }

  auto stage = [&](int k0, int buf) {
    // weight tile (512 bf16)
    for (int i = t; i < 512; i += 256) {
      int m = i >> 5, kk = i & 31, k = k0 + kk;
      sW[buf][i] = (k < K) ? wgt[(long)(mblock + m) * K + k] : bfz();
    }
    // im2col tile: 32 K-rows x 2 columns per thread
    __bf16* dst = &sI[buf][2 * t];
    for (int r = 0; r < 32; ++r) {
      int k = k0 + r;
      __bf16 v0 = bfz(), v1 = bfz();
      if (k < K) {
        int ci = k / 27, kk = k % 27;
        int kd = kk / 9, kh = (kk / 3) % 3, kw = kk % 3;
        const __bf16* base = act + (long)ci * Din * Hin * Win;
#pragma unroll
        for (int c = 0; c < 2; ++c) {
          int id, ih, iw; bool ok;
          if (!deconv) {
            id = od[c] * stride + kd - 1;
            ih = oh[c] * stride + kh - 1;
            iw = ow[c] * stride + kw - 1;
            ok = (unsigned)id < (unsigned)Din && (unsigned)ih < (unsigned)Hin &&
                 (unsigned)iw < (unsigned)Win;
          } else {
            int td = od[c] + kd - 1, th = oh[c] + kh - 1, tw = ow[c] + kw - 1;
            ok = !((td | th | tw) & 1);
            id = td >> 1; ih = th >> 1; iw = tw >> 1;
            ok = ok && (unsigned)id < (unsigned)Din && (unsigned)ih < (unsigned)Hin &&
                 (unsigned)iw < (unsigned)Win;
          }
          __bf16 v = ok ? base[((long)id * Hin + ih) * Win + iw] : bfz();
          if (c == 0) v0 = v; else v1 = v;
        }
      }
      // two adjacent bf16 -> one 32-bit LDS store
      unsigned pack = (unsigned)bf_bits(v0) | ((unsigned)bf_bits(v1) << 16);
      *(unsigned*)&dst[(long)r * LPITCH] = pack;
    }
  };

  v8f acc[4];
#pragma unroll
  for (int j = 0; j < 4; ++j) acc[j] = (v8f){0.f,0.f,0.f,0.f,0.f,0.f,0.f,0.f};

  stage(0, 0);
  __syncthreads();

  const int g = lane >> 4, mr = lane & 15;
  int cur = 0;
  for (int k0 = 0; k0 < K; k0 += 32) {
    if (t == 0 && k0 + 32 < K)
      __builtin_prefetch(&wgt[(long)mblock * K + k0 + 32], 0, 1);
    if (k0 + 32 < K)
      stage(k0 + 32, cur ^ 1);          // overlap with compute on buffer `cur`

    // A fragment (ISA 7.12.2: lane = M row, split-K element map)
    v16bf a;
#pragma unroll
    for (int e = 0; e < 16; ++e) {
      int vv = e >> 1, hh = e & 1;
      int kk = ((vv < 4) ? 0 : 16) + g * 8 + 2 * (vv & 3) + hh;
      a[e] = sW[cur][mr * 32 + kk];
    }
    // 4 x (B fragment + WMMA), B: lane = N column, element = K (split-K map)
    const __bf16* src = &sI[cur][wv * 64 + mr];
#pragma unroll
    for (int j = 0; j < 4; ++j) {
      v16bf b;
#pragma unroll
      for (int e = 0; e < 16; ++e) b[e] = src[(long)(((e < 8) ? 0 : 8) + g * 8 + e) * LPITCH + j * 16];
      acc[j] = __builtin_amdgcn_wmma_f32_16x16x32_bf16(
          false, a, false, b, (short)0, acc[j], false, false);
    }
    __syncthreads();
    cur ^= 1;
  }

  // D layout: N = lane%16 within sub-tile, M = 8*(lane/16) + r
#pragma unroll
  for (int j = 0; j < 4; ++j) {
    long nout = nblock + wv * 64 + j * 16 + (lane & 15);
#pragma unroll
    for (int r = 0; r < 8; ++r) {
      int m = mblock + (lane >> 4) * 8 + r;
      out[(long)m * NS + nout] = acc[j][r];
    }
  }
}

// ---------------------------------------------------------------------------
// training-mode BatchNorm3d: per-channel stats then fused apply
// ---------------------------------------------------------------------------
__global__ __launch_bounds__(256) void k_bn_stats(
    const float* __restrict__ x, float* __restrict__ mean,
    float* __restrict__ var, long S)
{
  int c = blockIdx.x;
  const float* p = x + (long)c * S;
  float s = 0.f, q = 0.f;
  for (long i = threadIdx.x; i < S; i += blockDim.x) {
    float v = p[i]; s += v; q += v * v;
  }
  __shared__ float ss[256], qq[256];
  ss[threadIdx.x] = s; qq[threadIdx.x] = q; __syncthreads();
  for (int str = 128; str > 0; str >>= 1) {
    if (threadIdx.x < str) {
      ss[threadIdx.x] += ss[threadIdx.x + str];
      qq[threadIdx.x] += qq[threadIdx.x + str];
    }
    __syncthreads();
  }
  if (threadIdx.x == 0) {
    float m = ss[0] / (float)S;
    mean[c] = m;
    var[c]  = qq[0] / (float)S - m * m;
  }
}

__global__ __launch_bounds__(256) void k_bn_apply(
    const float* __restrict__ x, const float* __restrict__ mean,
    const float* __restrict__ var, const float* __restrict__ gamma,
    const float* __restrict__ beta, const float* __restrict__ resid,
    float* __restrict__ outf, __bf16* __restrict__ outb, long S, int relu)
{
  int c = blockIdx.y;
  float sc = gamma[c] * rsqrtf(var[c] + 1e-5f);
  float sh = beta[c] - mean[c] * sc;
  long i = (long)blockIdx.x * blockDim.x + threadIdx.x;
  if (i >= S) return;
  long idx = (long)c * S + i;
  float v = x[idx] * sc + sh;
  if (resid) v += resid[idx];
  if (relu)  v = fmaxf(v, 0.f);
  if (outf)  outf[idx] = v;
  if (outb)  outb[idx] = f2bf(v);
}

// ---------------------------------------------------------------------------
// SGA guidance: L1-normalize the 5 weights per (dir, channel, pixel)
// g layout: [dir*5*C + c*5 + j][H*W]; kn keeps the same layout.
// ---------------------------------------------------------------------------
__global__ void k_guidance(const float* __restrict__ g, float* __restrict__ kn,
                           int C, long HW)
{
  long tid = (long)blockIdx.x * blockDim.x + threadIdx.x;
  long total = 4L * C * HW;
  if (tid >= total) return;
  long pix = tid % HW;
  long rest = tid / HW;
  int c = (int)(rest % C);
  int i = (int)(rest / C);
  const float* gp = g + ((long)(i * 5 * C + c * 5)) * HW + pix;
  float a0 = gp[0], a1 = gp[HW], a2 = gp[2 * HW], a3 = gp[3 * HW], a4 = gp[4 * HW];
  float s  = fabsf(a0) + fabsf(a1) + fabsf(a2) + fabsf(a3) + fabsf(a4);
  float inv = 1.f / fmaxf(s, 1e-12f);
  float* kp = kn + ((long)(i * 5 * C + c * 5)) * HW + pix;
  kp[0] = a0 * inv; kp[HW] = a1 * inv; kp[2 * HW] = a2 * inv;
  kp[3 * HW] = a3 * inv; kp[4 * HW] = a4 * inv;
}

// ---------------------------------------------------------------------------
// SGA directional scan. Recurrence state (one D-vector per line) lives in
// LDS; threadIdx.x = d (padded to pow2), threadIdx.y = line within block.
// axis 0: scan along W (line = h) ; axis 1: scan along H (line = w).
// accum: max-accumulate into out (folds the 4-direction max).
// ---------------------------------------------------------------------------
__global__ __launch_bounds__(512) void k_scan(
    const float* __restrict__ x, const float* __restrict__ kn,
    float* __restrict__ out, int D, int H, int W, int axis, int rev, int accum)
{
  const int DP = blockDim.x;                 // 32 or 64 (pow2 >= D)
  const int c  = blockIdx.x;
  const int y  = blockIdx.y * blockDim.y + threadIdx.y; // h (axis0) / w (axis1)
  const int d  = threadIdx.x;
  const long HW = (long)H * W;
  const int  L  = (axis == 0) ? W : H;

  __shared__ float prev[512];
  __shared__ float red[512];
  const int  base = threadIdx.y * DP + d;
  const bool act  = (d < D);

  const float* xc = x  + ((long)c * D) * HW;
  const float* kc = kn + ((long)c * 5) * HW;
  float* op = out + ((long)c * D) * HW;

  float o = 0.f;
  for (int t = 0; t < L; ++t) {
    int s = rev ? (L - 1 - t) : t;
    int h = (axis == 0) ? y : s;
    int w = (axis == 0) ? s : y;
    long pix = (long)h * W + w;
    float xv = act ? xc[(long)d * HW + pix] : 0.f;

    if (t == 0) {
      o = xv;                              // boundary: aggregated == raw
    } else {
      red[base] = act ? prev[base] : -3.0e38f;
      __syncthreads();
      for (int str = DP >> 1; str > 0; str >>= 1) {
        if (d < str) red[base] = fmaxf(red[base], red[base + str]);
        __syncthreads();
      }
      float pmax = red[threadIdx.y * DP];
      float p   = act ? prev[base] : 0.f;
      float pdm = (act && d > 0)     ? prev[base - 1] : 0.f;
      float pdp = (act && d < D - 1) ? prev[base + 1] : 0.f;
      float k0 = kc[pix], k1 = kc[HW + pix], k2 = kc[2 * HW + pix],
            k3 = kc[3 * HW + pix], k4 = kc[4 * HW + pix];
      o = k0 * xv + k1 * p + k2 * pdm + k3 * pdp + k4 * pmax;
    }
    __syncthreads();
    prev[base] = o;
    __syncthreads();
    if (act) {
      long oi = (long)d * HW + pix;
      if (accum) op[oi] = fmaxf(op[oi], o);
      else       op[oi] = o;
    }
  }
}

// ---------------------------------------------------------------------------
// host-side helpers
// ---------------------------------------------------------------------------
static void conv(hipStream_t s, const __bf16* act, const __bf16* wgt, float* out,
                 int Cin, int Cout, int Din, int Hin, int Win,
                 int Dout, int Hout, int Wout, int stride, int deconv)
{
  long NS = (long)Dout * Hout * Wout;
  dim3 grid((unsigned)(NS / NB), (unsigned)(Cout / 16));
  k_conv_igemm<<<grid, 256, 0, s>>>(act, wgt, out, Cin, Din, Hin, Win,
                                    Dout, Hout, Wout, stride, deconv);
}

static void bn(hipStream_t s, const float* x, int C, long S,
               const float* g, const float* b, const float* resid,
               float* outf, __bf16* outb, int relu, float* mv)
{
  k_bn_stats<<<C, 256, 0, s>>>(x, mv, mv + 128, S);
  dim3 grid((unsigned)((S + 255) / 256), (unsigned)C);
  k_bn_apply<<<grid, 256, 0, s>>>(x, mv, mv + 128, g, b, resid, outf, outb, S, relu);
}

static void sga_block(hipStream_t s, float* x, const float* guid,
                      int C, int D, int H, int W,
                      const float* bn1g, const float* bn1b, const __bf16* wrefbf,
                      const float* bn2g, const float* bn2b,
                      float* Sb, float* KN, __bf16* BF, float* Fconv, float* mv)
{
  long HW = (long)H * W;
  long SS = (long)D * HW;
  long tot = 4L * C * HW;
  k_guidance<<<(unsigned)((tot + 255) / 256), 256, 0, s>>>(guid, KN, C, HW);
  int DP = (D <= 32) ? 32 : 64;
  int YT = 512 / DP;
  dim3 blk((unsigned)DP, (unsigned)YT);
  dim3 gw((unsigned)C, (unsigned)(H / YT));   // W scans: line = h
  dim3 gh((unsigned)C, (unsigned)(W / YT));   // H scans: line = w
  long kstride = 5L * C * HW;
  k_scan<<<gw, blk, 0, s>>>(x, KN + 0 * kstride, Sb, D, H, W, 0, 0, 0);
  k_scan<<<gw, blk, 0, s>>>(x, KN + 1 * kstride, Sb, D, H, W, 0, 1, 1);
  k_scan<<<gh, blk, 0, s>>>(x, KN + 2 * kstride, Sb, D, H, W, 1, 0, 1);
  k_scan<<<gh, blk, 0, s>>>(x, KN + 3 * kstride, Sb, D, H, W, 1, 1, 1);
  // relu(bn1(sga)) in-place on Sb, bf16 copy for the refine conv
  bn(s, Sb, C, SS, bn1g, bn1b, nullptr, Sb, BF, 1, mv);
  // refine conv C->C
  conv(s, BF, wrefbf, Fconv, C, C, D, H, W, D, H, W, 1, 0);
  // relu(bn2(conv) + rem) in-place on Fconv; bf16 copy for next conv
  bn(s, Fconv, C, SS, bn2g, bn2b, x, Fconv, BF, 1, mv);
}

// ---------------------------------------------------------------------------
extern "C" void kernel_launch(void* const* d_in, const int* in_sizes, int n_in,
                              void* d_out, int out_size, void* d_ws, size_t ws_size,
                              hipStream_t stream)
{
  (void)in_sizes; (void)n_in; (void)out_size; (void)ws_size;

  const float* x   = (const float*)d_in[0];
  const float* g1  = (const float*)d_in[1];
  const float* g11 = (const float*)d_in[2];
  const float* g12 = (const float*)d_in[3];
  const float* g2  = (const float*)d_in[4];
  auto P = [&](int i) { return (const float*)d_in[i]; };

  const int D0 = 48, H0 = 64, W0 = 128;
  const int D1 = 24, H1 = 32, W1 = 64;
  const int D2 = 12, H2 = 16, W2 = 32;
  const long SP0 = (long)D0 * H0 * W0;   // 393216
  const long SP1 = (long)D1 * H1 * W1;   // 49152
  const long SP2 = (long)D2 * H2 * W2;   // 6144

  // ---- workspace bump allocator (~345 MB) ----
  char* wsb = (char*)d_ws;
  size_t off = 0;
  auto alloc = [&](size_t bytes) -> void* {
    void* p = wsb + off;
    off = (off + bytes + 255) & ~(size_t)255;
    return p;
  };
  __bf16* xbf = (__bf16*)alloc(64 * SP0 * 2);
  float*  F0  = (float*)alloc(32 * SP0 * 4);     // rem0 / sga conv buffers
  float*  F1  = (float*)alloc(32 * SP0 * 4);
  float*  Sb  = (float*)alloc(32 * SP0 * 4);     // scan/max accumulate
  float*  KN  = (float*)alloc(4L * 5 * 32 * H0 * W0 * 4);
  __bf16* BF0 = (__bf16*)alloc(64 * SP0 * 2);    // bf16 staging (largest need)
  __bf16* BF1 = (__bf16*)alloc(96 * SP1 * 2);    // half-res concat staging
  float*  G0  = (float*)alloc(48 * SP1 * 4);
  float*  G1  = (float*)alloc(48 * SP1 * 4);
  float*  G2  = (float*)alloc(48 * SP1 * 4);
  float*  G3  = (float*)alloc(48 * SP1 * 4);
  float*  G4  = (float*)alloc(48 * SP1 * 4);
  float*  G5  = (float*)alloc(48 * SP1 * 4);
  float*  Q0  = (float*)alloc(64 * SP2 * 4);
  float*  mv  = (float*)alloc(256 * 4);          // mean[128] | var[128]
  __bf16* wb  = (__bf16*)alloc(2u << 20);        // bf16 weight pool

  // ---- weights -> bf16 (layout [Cout][Cin*27] is already contiguous) ----
  size_t woff = 0;
  auto wcvt = [&](const float* src, long ne) -> __bf16* {
    __bf16* dst = wb + woff;
    woff += (size_t)((ne + 127) & ~127L);
    k_tobf<<<(unsigned)((ne + 255) / 256), 256, 0, stream>>>(src, dst, ne);
    return dst;
  };
  __bf16* W_start = wcvt(P(5),  32L * 64 * 27);
  __bf16* W_sga1  = wcvt(P(10), 32L * 32 * 27);
  __bf16* W_sga11 = wcvt(P(15), 48L * 48 * 27);
  __bf16* W_sga12 = wcvt(P(20), 48L * 48 * 27);
  __bf16* W_sga2  = wcvt(P(25), 32L * 32 * 27);
  __bf16* W_c1a   = wcvt(P(28), 48L * 32 * 27);
  __bf16* W_c2a   = wcvt(P(31), 64L * 48 * 27);
  __bf16* W_d2a1  = wcvt(P(34), 48L * 64 * 27);
  __bf16* W_d2a2  = wcvt(P(37), 48L * 96 * 27);
  __bf16* W_d1a1  = wcvt(P(40), 32L * 48 * 27);
  __bf16* W_d1a2  = wcvt(P(43), 32L * 64 * 27);
  __bf16* W_end   = wcvt(P(46), 64L * 32 * 27);

  // ---- pipeline ----
  k_tobf<<<(unsigned)((64 * SP0 + 255) / 256), 256, 0, stream>>>(x, xbf, 64 * SP0);

  // conv_start (64->32) + BN (no relu)
  conv(stream, xbf, W_start, F0, 64, 32, D0, H0, W0, D0, H0, W0, 1, 0);
  bn(stream, F0, 32, SP0, P(6), P(7), nullptr, F1, nullptr, 0, mv);

  // sga1 (C=32, full res): out f32 = F0 (= rem0), out bf16 = BF0
  sga_block(stream, F1, g1, 32, D0, H0, W0, P(8), P(9), W_sga1, P(11), P(12),
            Sb, KN, BF0, F0, mv);

  // conv1a (32->48, stride 2) + BN + relu
  conv(stream, BF0, W_c1a, G0, 32, 48, D0, H0, W0, D1, H1, W1, 2, 0);
  bn(stream, G0, 48, SP1, P(29), P(30), nullptr, G1, nullptr, 1, mv);

  // sga11 (C=48, half res): out f32 = G2 (= rem1), out bf16 = BF0
  sga_block(stream, G1, g11, 48, D1, H1, W1, P(13), P(14), W_sga11, P(16), P(17),
            Sb, KN, BF0, G2, mv);

  // conv2a (48->64, stride 2) + BN + relu (bf16 for deconv)
  conv(stream, BF0, W_c2a, Q0, 48, 64, D1, H1, W1, D2, H2, W2, 2, 0);
  bn(stream, Q0, 64, SP2, P(32), P(33), nullptr, Q0, BF0, 1, mv);

  // deconv2a part1 (64->48 upsample) + BN + relu -> BF1[0:48]
  conv(stream, BF0, W_d2a1, G3, 64, 48, D2, H2, W2, D1, H1, W1, 1, 1);
  bn(stream, G3, 48, SP1, P(35), P(36), nullptr, nullptr, BF1, 1, mv);
  // concat rem1 -> BF1[48:96]
  k_tobf<<<(unsigned)((48 * SP1 + 255) / 256), 256, 0, stream>>>(G2, BF1 + 48 * SP1, 48 * SP1);
  // deconv2a part2 (96->48) + BN + relu
  conv(stream, BF1, W_d2a2, G4, 96, 48, D1, H1, W1, D1, H1, W1, 1, 0);
  bn(stream, G4, 48, SP1, P(38), P(39), nullptr, G4, nullptr, 1, mv);

  // sga12 (C=48): out f32 = G5, out bf16 = BF0
  sga_block(stream, G4, g12, 48, D1, H1, W1, P(18), P(19), W_sga12, P(21), P(22),
            Sb, KN, BF0, G5, mv);

  // deconv1a part1 (48->32 upsample) + BN + relu -> BF0[0:32]
  conv(stream, BF0, W_d1a1, F1, 48, 32, D1, H1, W1, D0, H0, W0, 1, 1);
  bn(stream, F1, 32, SP0, P(41), P(42), nullptr, nullptr, BF0, 1, mv);
  // concat rem0 -> BF0[32:64]
  k_tobf<<<(unsigned)((32 * SP0 + 255) / 256), 256, 0, stream>>>(F0, BF0 + 32 * SP0, 32 * SP0);
  // deconv1a part2 (64->32) + BN (no relu)
  conv(stream, BF0, W_d1a2, F1, 64, 32, D0, H0, W0, D0, H0, W0, 1, 0);
  bn(stream, F1, 32, SP0, P(44), P(45), nullptr, F1, nullptr, 0, mv);

  // sga2 (C=32): out f32 = F0, out bf16 = BF0
  sga_block(stream, F1, g2, 32, D0, H0, W0, P(23), P(24), W_sga2, P(26), P(27),
            Sb, KN, BF0, F0, mv);

  // conv_end (32->64) + relu -> d_out
  float* out = (float*)d_out;
  conv(stream, BF0, W_end, out, 32, 64, D0, H0, W0, D0, H0, W0, 1, 0);
  k_relu<<<(unsigned)((64 * SP0 + 255) / 256), 256, 0, stream>>>(out, 64 * SP0);
}
